// PMA_10677288698629
// MI455X (gfx1250) — matrix operand order repaired and checked by reading, not compiled, not run on hardware
//
#include <hip/hip_runtime.h>
#include <hip/hip_bf16.h>

// ---------------------------------------------------------------------------
// Types for CDNA5 WMMA (wave32, 16x16x32 bf16 -> f32 accumulate)
// ---------------------------------------------------------------------------
typedef __attribute__((ext_vector_type(16))) __bf16       v16bf;
typedef __attribute__((ext_vector_type(8)))  float        v8f;
typedef __attribute__((ext_vector_type(8)))  unsigned int u32x8;
typedef __attribute__((__vector_size__(4 * sizeof(int)))) int v4i;

#define DIM   256
#define HEADS 4
#define HD    64
#define GRAPHS 4096
#define FFN   1024

// gfx1250 async global->LDS path (ASYNCcnt), with synchronous fallback
#if defined(__has_builtin)
#if __has_builtin(__builtin_amdgcn_global_load_async_to_lds_b128)
#define HAVE_ASYNC_LDS 1
#endif
#endif
#ifndef HAVE_ASYNC_LDS
#define HAVE_ASYNC_LDS 0
#endif

__device__ __forceinline__ unsigned short f32_to_bf16(float f) {
  unsigned int u = __builtin_bit_cast(unsigned int, f);
  u += 0x7fffu + ((u >> 16) & 1u);           // round-to-nearest-even
  return (unsigned short)(u >> 16);
}

__device__ __forceinline__ v8f wmma_bf16(v16bf a, v16bf b, v8f c) {
  return __builtin_amdgcn_wmma_f32_16x16x32_bf16(
      /*neg_a=*/false, a, /*neg_b=*/false, b,
      /*c_mod=*/(short)0, c, /*reuse_a=*/false, /*reuse_b=*/false);
}

// A-fragment (16x32 bf16) from a row-major bf16 LDS matrix.
// ISA layout: lane m=lane&15, half=lane>>4; VGPR v in 0..3 -> K = half*8+2v,
// VGPR v in 4..7 -> K = 16 + half*8 + 2(v-4); each VGPR packs K,K+1.
__device__ __forceinline__ v16bf load_a_frag_lds(const unsigned short* A,
                                                 int stride, int koff, int lane) {
  const int m = lane & 15, half = lane >> 4;
  const unsigned short* row = A + m * stride + koff;
  u32x8 d;
#pragma unroll
  for (int v = 0; v < 4; ++v) {
    d[v]     = *(const unsigned int*)(row + (half * 8 + 2 * v));
    d[v + 4] = *(const unsigned int*)(row + (16 + half * 8 + 2 * v));
  }
  return __builtin_bit_cast(v16bf, d);
}

// B-fragment: pre-packed by pack_b into per-lane contiguous 32B chunks.
__device__ __forceinline__ v16bf load_b_frag(const unsigned int* pack,
                                             int tile, int lane) {
  const u32x8 d = *(const u32x8*)(pack + ((size_t)tile * 32 + lane) * 8);
  return __builtin_bit_cast(v16bf, d);
}

// ---------------------------------------------------------------------------
// Kernel 1: fold q = seed@Wq^T + bq into per-head score vectors
//   p[h][c] = (1/sqrt(hd)) * sum_{j in head h} q[j] * Wk[j][c]
// ---------------------------------------------------------------------------
__global__ void prep_p(const float* __restrict__ seed,
                       const float* __restrict__ ipw,
                       const float* __restrict__ ipb,
                       float* __restrict__ p_ws) {
  __shared__ float q[DIM];
  const int t = threadIdx.x;
  float acc = ipb[t];
  const float* w = ipw + (size_t)t * DIM;
  for (int c = 0; c < DIM; ++c) acc += seed[c] * w[c];
  q[t] = acc;
  __syncthreads();
  float pr[HEADS] = {0.f, 0.f, 0.f, 0.f};
  for (int j = 0; j < DIM; ++j)
    pr[j >> 6] += q[j] * ipw[(size_t)(DIM + j) * DIM + t];
#pragma unroll
  for (int hh = 0; hh < HEADS; ++hh) p_ws[hh * DIM + t] = 0.125f * pr[hh];
}

// ---------------------------------------------------------------------------
// Kernel 2: segment offsets of sorted batch (int64): off[g] = lower_bound(g)
// ---------------------------------------------------------------------------
__global__ void seg_offsets(const long long* __restrict__ batch, int N, int G,
                            int* __restrict__ off) {
  const int g = blockIdx.x * blockDim.x + threadIdx.x;
  if (g > G) return;
  int lo = 0, hi = N;
  while (lo < hi) {
    const int mid = (lo + hi) >> 1;
    if (batch[mid] < (long long)g) lo = mid + 1; else hi = mid;
  }
  off[g] = lo;
}

// ---------------------------------------------------------------------------
// Kernel 3: pack weight W[n][k] (f32) into bf16 WMMA B-fragments.
// dst dword layout: [(nt*ktiles+kt)*32 + lane]*8 + v, value = {W[n][k],W[n][k+1]}
// with n = nt*16 + (lane&15), k = kt*32 + (v>>2)*16 + (lane>>4)*8 + 2(v&3).
// ---------------------------------------------------------------------------
__global__ void pack_b(const float* __restrict__ W, int Nrows, int K,
                       unsigned int* __restrict__ dst) {
  const int total = (Nrows / 16) * (K / 32) * 32 * 8;
  const int idx = blockIdx.x * blockDim.x + threadIdx.x;
  if (idx >= total) return;
  const int v = idx & 7;
  const int lane = (idx >> 3) & 31;
  const int tile = idx >> 8;
  const int ktiles = K / 32;
  const int nt = tile / ktiles, kt = tile % ktiles;
  const int n = nt * 16 + (lane & 15);
  const int k = kt * 32 + (v >> 2) * 16 + (lane >> 4) * 8 + 2 * (v & 3);
  const unsigned int lo = f32_to_bf16(W[(size_t)n * K + k]);
  const unsigned int hi = f32_to_bf16(W[(size_t)n * K + k + 1]);
  dst[idx] = lo | (hi << 16);
}

// ---------------------------------------------------------------------------
// Kernel 4: main streaming pass. One block per graph, 256 threads (8 waves).
// Online softmax over the sorted segment; accumulates Ahat[g][h][c] =
// segment_sum(e[n,h]*h[n,c]) / denom[h].  Memory-bound: reads h exactly once,
// double-buffered via GLOBAL_LOAD_ASYNC_TO_LDS_B128 (ASYNCcnt) when available.
// ---------------------------------------------------------------------------

// Stage one 32-node chunk into an LDS buffer. Each wave owns 4 nodes and
// issues exactly 8 async b128 ops (pad rows clamp the address; never read).
__device__ __forceinline__ void stage_chunk(float* __restrict__ buf,
                                            const float* __restrict__ h,
                                            int base, int n1, int wv, int lane) {
#pragma unroll
  for (int i = 0; i < 4; ++i) {
    const int nl = wv * 4 + i;
    int n = base + nl;
    n = (n < n1) ? n : (n1 - 1);     // clamp pads (segment is non-empty)
    const float* src = h + (size_t)n * DIM + lane * 8;
    float* dst = buf + nl * DIM + lane * 8;
#if HAVE_ASYNC_LDS
    __builtin_amdgcn_global_load_async_to_lds_b128(
        (v4i*)src,
        (__attribute__((address_space(3))) v4i*)(unsigned int)(size_t)dst,
        0, 0);
    __builtin_amdgcn_global_load_async_to_lds_b128(
        (v4i*)(src + 4),
        (__attribute__((address_space(3))) v4i*)(unsigned int)(size_t)(dst + 4),
        0, 0);
#else
    const float4* sp = (const float4*)src;
    const float4 a0 = sp[0], a1 = sp[1];
    float4* dp = (float4*)dst;
    dp[0] = a0; dp[1] = a1;
#endif
  }
}

__device__ __forceinline__ void wait_async_le8() {
#if HAVE_ASYNC_LDS
  asm volatile("s_wait_asynccnt 0x8" ::: "memory");
#endif
}
__device__ __forceinline__ void wait_async_0() {
#if HAVE_ASYNC_LDS
  asm volatile("s_wait_asynccnt 0x0" ::: "memory");
#endif
}

__global__ void pma_main(const float* __restrict__ h,
                         const int* __restrict__ off,
                         const float* __restrict__ p_ws,
                         float* __restrict__ ahat) {
  __shared__ __align__(16) float hch[2 * 32 * DIM];  // double-buffered chunks (64KB)
  __shared__ __align__(16) float sc[32][4];          // chunk scores
  __shared__ __align__(16) float e_lds[32][4];       // chunk exp weights
  __shared__ float m_s[4], s_s[4], alpha_s[4];

  const int g = blockIdx.x;
  const int tid = threadIdx.x;
  const int lane = tid & 31, wv = tid >> 5;
  const int n0 = off[g], n1 = off[g + 1];
  const int nchunks = (n1 - n0 + 31) >> 5;

  // per-head score vectors for this lane's 8 channels
  float preg[HEADS][8];
#pragma unroll
  for (int hh = 0; hh < HEADS; ++hh)
#pragma unroll
    for (int j = 0; j < 8; ++j) preg[hh][j] = p_ws[hh * DIM + lane * 8 + j];

  if (tid < 4) { m_s[tid] = -__builtin_inff(); s_s[tid] = 0.f; }
  float acc[HEADS] = {0.f, 0.f, 0.f, 0.f};
  __syncthreads();

  if (nchunks > 0) stage_chunk(hch, h, n0, n1, wv, lane);

  for (int c = 0; c < nchunks; ++c) {
    const int base = n0 + c * 32;
    const int lim = min(32, n1 - base);
    float* cur = hch + (c & 1) * (32 * DIM);

    if (c + 1 < nchunks) {
      stage_chunk(hch + ((c + 1) & 1) * (32 * DIM), h, base + 32, n1, wv, lane);
      wait_async_le8();                 // chunk c complete (in-order)
    } else {
      wait_async_0();
    }
#if !HAVE_ASYNC_LDS
    if (c + 2 < nchunks) {              // confirmed: lowers to global_prefetch_b8
      const int pn = min(base + 64 + wv * 4, n1 - 1);
      __builtin_prefetch(h + (size_t)pn * DIM + lane * 8, 0, 1);
    }
#endif
    __syncthreads();                    // chunk c visible to all waves

    // ---- Phase A: scores from LDS (wave-wide dot + shfl reduction) --------
#pragma unroll
    for (int i = 0; i < 4; ++i) {
      const int nl = wv * 4 + i;
      if (nl >= lim) break;             // uniform per wave-iteration
      const float4* lp = (const float4*)&cur[nl * DIM + lane * 8];
      float av[8];
      *(float4*)&av[0] = lp[0];
      *(float4*)&av[4] = lp[1];
      float pp[HEADS] = {0.f, 0.f, 0.f, 0.f};
#pragma unroll
      for (int hh = 0; hh < HEADS; ++hh)
#pragma unroll
        for (int j = 0; j < 8; ++j) pp[hh] += av[j] * preg[hh][j];
#pragma unroll
      for (int o = 16; o > 0; o >>= 1)
#pragma unroll
        for (int hh = 0; hh < HEADS; ++hh) pp[hh] += __shfl_xor(pp[hh], o);
      if (lane == 0)
#pragma unroll
        for (int hh = 0; hh < HEADS; ++hh) sc[nl][hh] = pp[hh];
    }
    __syncthreads();

    // ---- Phase B: online softmax update (4 lanes, one per head) -----------
    if (tid < 4) {
      const int hh = tid;
      float cm = -__builtin_inff();
      for (int n = 0; n < lim; ++n) cm = fmaxf(cm, sc[n][hh]);
      const float mo = m_s[hh];
      const float nm = fmaxf(mo, cm);
      const float al = __expf(mo - nm);  // exp(-inf)=0 on first chunk
      float ssum = 0.f;
      for (int n = 0; n < lim; ++n) {
        const float e = __expf(sc[n][hh] - nm);
        e_lds[n][hh] = e;
        ssum += e;
      }
      s_s[hh] = s_s[hh] * al + ssum;
      m_s[hh] = nm;
      alpha_s[hh] = al;
    }
    __syncthreads();

    // ---- Phase C: rescale + accumulate A[h, c=tid] -------------------------
    {
      const float a0 = alpha_s[0], a1 = alpha_s[1], a2 = alpha_s[2], a3 = alpha_s[3];
      acc[0] *= a0; acc[1] *= a1; acc[2] *= a2; acc[3] *= a3;
      for (int n = 0; n < lim; ++n) {
        const float val = cur[n * DIM + tid];
        const float4 e4 = *(const float4*)&e_lds[n][0];
        acc[0] += e4.x * val; acc[1] += e4.y * val;
        acc[2] += e4.z * val; acc[3] += e4.w * val;
      }
    }
    __syncthreads();                    // cur free for reuse by chunk c+2
  }

  // ---- write Ahat = A / max(denom,1e-9); empty graph -> 0 ------------------
#pragma unroll
  for (int hh = 0; hh < HEADS; ++hh) {
    const float s = s_s[hh];
    const float inv = (s > 0.f) ? (1.f / fmaxf(s, 1e-9f)) : 0.f;
    ahat[(size_t)g * (HEADS * DIM) + hh * DIM + tid] = acc[hh] * inv;
  }
}

// ---------------------------------------------------------------------------
// Kernel 5: fused WMMA tail for a tile of 16 graphs:
//   attn = Ahat @ Wv_h^T  ->  o = attn@out_w^T + out_b + seed  ->  LN1
//   f = relu(x1@w1^T+b1)@w2^T+b2  ->  LN2(x1+f)  ->  mask empty graphs
// ---------------------------------------------------------------------------
__global__ void pma_tail(const float* __restrict__ ahat,
                         const int* __restrict__ off,
                         const float* __restrict__ seed,
                         const float* __restrict__ out_b,
                         const float* __restrict__ b1,
                         const float* __restrict__ b2,
                         const float* __restrict__ g1v, const float* __restrict__ be1,
                         const float* __restrict__ g2v, const float* __restrict__ be2,
                         const unsigned int* __restrict__ wvp,
                         const unsigned int* __restrict__ owp,
                         const unsigned int* __restrict__ w1p,
                         const unsigned int* __restrict__ w2p,
                         float* __restrict__ out) {
  __shared__ __align__(16) unsigned short s_r0[16 * FFN];  // 32KB: ahat_bf16 | f1_bf16 | (f32) x2pre
  __shared__ __align__(16) unsigned short s_r1[16 * DIM];  // 8KB : attn_bf16 | x1_bf16
  __shared__ __align__(16) float          s_x [16 * DIM];  // 16KB: residual (f32)

  const int tid = threadIdx.x;
  const int lane = tid & 31, wv = tid >> 5;
  const int g0 = blockIdx.x * 16;
  const int mrow = lane & 15, half = lane >> 4;

  // ---- Stage 0: Ahat[16][4*256] f32 -> bf16 LDS (coalesced) ---------------
#pragma unroll
  for (int i = 0; i < 64; ++i) {
    const int idx = tid + i * 256;
    s_r0[idx] = f32_to_bf16(ahat[(size_t)g0 * (HEADS * DIM) + idx]);
  }
  __syncthreads();

  // ---- Stage 1: attn = Ahat_h @ Wv_h^T  (16 col tiles, 2 per wave) --------
#pragma unroll
  for (int t = 0; t < 2; ++t) {
    const int nt = wv * 2 + t;        // output col tile (0..15)
    const int head = nt >> 2;         // 4 tiles per head
    v8f c = {0.f, 0.f, 0.f, 0.f, 0.f, 0.f, 0.f, 0.f};
#pragma unroll
    for (int kt = 0; kt < 8; ++kt) {
      const v16bf a = load_a_frag_lds(s_r0, FFN, head * DIM + kt * 32, lane);
      const v16bf b = load_b_frag(wvp, nt * 8 + kt, lane);
      c = wmma_bf16(a, b, c);
    }
    const int col = nt * 16 + mrow;
#pragma unroll
    for (int r = 0; r < 8; ++r)
      s_r1[(r + 8 * half) * DIM + col] = f32_to_bf16(c[r]);
  }
  __syncthreads();

  // ---- Stage 2: o = attn @ out_w^T + out_b + seed -> s_x (f32) ------------
#pragma unroll
  for (int t = 0; t < 2; ++t) {
    const int nt = wv * 2 + t;
    v8f c = {0.f, 0.f, 0.f, 0.f, 0.f, 0.f, 0.f, 0.f};
#pragma unroll
    for (int kt = 0; kt < 8; ++kt) {
      const v16bf a = load_a_frag_lds(s_r1, DIM, kt * 32, lane);
      const v16bf b = load_b_frag(owp, nt * 8 + kt, lane);
      c = wmma_bf16(a, b, c);
    }
    const int col = nt * 16 + mrow;
    const float addv = out_b[col] + seed[col];
#pragma unroll
    for (int r = 0; r < 8; ++r)
      s_x[(r + 8 * half) * DIM + col] = c[r] + addv;
  }
  __syncthreads();

  // ---- Stage 3: LN1 (f32) -> s_x (post-LN f32) and s_r1 (bf16) ------------
#pragma unroll
  for (int rr = 0; rr < 2; ++rr) {
    const int row = wv * 2 + rr;
    float vals[8], s = 0.f, sq = 0.f;
#pragma unroll
    for (int j = 0; j < 8; ++j) {
      vals[j] = s_x[row * DIM + lane * 8 + j];
      s += vals[j]; sq += vals[j] * vals[j];
    }
#pragma unroll
    for (int o = 16; o > 0; o >>= 1) { s += __shfl_xor(s, o); sq += __shfl_xor(sq, o); }
    const float mu = s * (1.f / DIM);
    const float var = sq * (1.f / DIM) - mu * mu;
    const float rstd = rsqrtf(var + 1e-5f);
#pragma unroll
    for (int j = 0; j < 8; ++j) {
      const int col = lane * 8 + j;
      const float xv = (vals[j] - mu) * rstd * g1v[col] + be1[col];
      s_x[row * DIM + col] = xv;
      s_r1[row * DIM + col] = f32_to_bf16(xv);
    }
  }
  __syncthreads();

  // ---- Stage 4: f1 = relu(x1 @ w1^T + b1) -> s_r0 bf16 [16][1024] ---------
#pragma unroll
  for (int t = 0; t < 8; ++t) {
    const int nt = wv * 8 + t;        // 0..63
    v8f c = {0.f, 0.f, 0.f, 0.f, 0.f, 0.f, 0.f, 0.f};
#pragma unroll
    for (int kt = 0; kt < 8; ++kt) {
      const v16bf a = load_a_frag_lds(s_r1, DIM, kt * 32, lane);
      const v16bf b = load_b_frag(w1p, nt * 8 + kt, lane);
      c = wmma_bf16(a, b, c);
    }
    const int col = nt * 16 + mrow;
    const float bb = b1[col];
#pragma unroll
    for (int r = 0; r < 8; ++r)
      s_r0[(r + 8 * half) * FFN + col] = f32_to_bf16(fmaxf(c[r] + bb, 0.f));
  }
  __syncthreads();

  // ---- Stage 5: f2 = f1 @ w2^T + b2 + x1 -> stage into s_r0 as f32 --------
  v8f cc[2];
#pragma unroll
  for (int t = 0; t < 2; ++t) {
    const int nt = wv * 2 + t;
    v8f c = {0.f, 0.f, 0.f, 0.f, 0.f, 0.f, 0.f, 0.f};
    for (int kt = 0; kt < 32; ++kt) {
      const v16bf a = load_a_frag_lds(s_r0, FFN, kt * 32, lane);
      const v16bf b = load_b_frag(w2p, nt * 32 + kt, lane);
      c = wmma_bf16(a, b, c);
    }
    cc[t] = c;
  }
  __syncthreads();   // all waves done reading f1 before overwrite
  {
    float* st = (float*)s_r0;        // reuse region as f32 [16][256]
#pragma unroll
    for (int t = 0; t < 2; ++t) {
      const int nt = wv * 2 + t;
      const int col = nt * 16 + mrow;
      const float bb = b2[col];
#pragma unroll
      for (int r = 0; r < 8; ++r) {
        const int m = r + 8 * half;
        st[m * DIM + col] = cc[t][r] + bb + s_x[m * DIM + col];
      }
    }
  }
  __syncthreads();

  // ---- Stage 6: LN2 + empty-graph mask -> global out ----------------------
  {
    const float* st = (const float*)s_r0;
#pragma unroll
    for (int rr = 0; rr < 2; ++rr) {
      const int row = wv * 2 + rr;
      const int g = g0 + row;
      const float mask = (off[g + 1] - off[g] > 0) ? 1.f : 0.f;
      float vals[8], s = 0.f, sq = 0.f;
#pragma unroll
      for (int j = 0; j < 8; ++j) {
        vals[j] = st[row * DIM + lane * 8 + j];
        s += vals[j]; sq += vals[j] * vals[j];
      }
#pragma unroll
      for (int o = 16; o > 0; o >>= 1) { s += __shfl_xor(s, o); sq += __shfl_xor(sq, o); }
      const float mu = s * (1.f / DIM);
      const float var = sq * (1.f / DIM) - mu * mu;
      const float rstd = rsqrtf(var + 1e-5f);
#pragma unroll
      for (int j = 0; j < 8; ++j) {
        const int col = lane * 8 + j;
        out[(size_t)g * DIM + col] =
            ((vals[j] - mu) * rstd * g2v[col] + be2[col]) * mask;
      }
    }
  }
}

// ---------------------------------------------------------------------------
extern "C" void kernel_launch(void* const* d_in, const int* in_sizes, int n_in,
                              void* d_out, int out_size, void* d_ws, size_t ws_size,
                              hipStream_t stream) {
  const float*     h     = (const float*)d_in[0];
  const long long* batch = (const long long*)d_in[1];
  const float*     seed  = (const float*)d_in[3];
  const float*     ipw   = (const float*)d_in[4];
  const float*     ipb   = (const float*)d_in[5];
  const float*     out_w = (const float*)d_in[6];
  const float*     out_b = (const float*)d_in[7];
  const float*     w1    = (const float*)d_in[8];
  const float*     b1    = (const float*)d_in[9];
  const float*     w2    = (const float*)d_in[10];
  const float*     b2    = (const float*)d_in[11];
  const float*     g1v   = (const float*)d_in[12];
  const float*     be1   = (const float*)d_in[13];
  const float*     g2v   = (const float*)d_in[14];
  const float*     be2   = (const float*)d_in[15];
  float* out = (float*)d_out;

  const int N = in_sizes[0] / DIM;
  const int G = GRAPHS;

  // workspace layout (byte offsets, all 32B-aligned)
  char* ws = (char*)d_ws;
  float*        p_ws = (float*)ws;                                   //   4 KB
  int*          off  = (int*)(ws + 4096);                            //  16.4 KB
  float*        ahat = (float*)(ws + 20480);                         //  16 MB
  unsigned int* wvp  = (unsigned int*)(ws + 20480 + (size_t)G * 1024 * 4);
  unsigned int* owp  = wvp + 32768;    // 128 KB each for 256x256
  unsigned int* w1p  = owp + 32768;    // 512 KB
  unsigned int* w2p  = w1p + 131072;   // 512 KB

  prep_p<<<1, 256, 0, stream>>>(seed, ipw, ipb, p_ws);
  seg_offsets<<<(G + 1 + 255) / 256, 256, 0, stream>>>(batch, N, G, off);
  pack_b<<<(32768 + 255) / 256, 256, 0, stream>>>(ipw + 512 * 256, 256, 256, wvp);
  pack_b<<<(32768 + 255) / 256, 256, 0, stream>>>(out_w, 256, 256, owp);
  pack_b<<<(131072 + 255) / 256, 256, 0, stream>>>(w1, 1024, 256, w1p);
  pack_b<<<(131072 + 255) / 256, 256, 0, stream>>>(w2, 256, 1024, w2p);
  pma_main<<<G, 256, 0, stream>>>(h, off, p_ws, ahat);
  pma_tail<<<G / 16, 256, 0, stream>>>(ahat, off, seed, out_b, b1, b2,
                                       g1v, be1, g2v, be2,
                                       wvp, owp, w1p, w2p, out);
}
